// HUpCrossAttn_9010841387183
// MI455X (gfx1250) — matrix operand order repaired and checked by reading, not compiled, not run on hardware
//
#include <hip/hip_runtime.h>

// ---------------------------------------------------------------------------
// MI455X (gfx1250) fused cross-attention upsample.
// Roofline: ~288 MB min traffic -> ~12.4 us @ 23.3 TB/s; 25.6 GFLOP of
// matmul must run > 2 PFLOP/s to stay under that floor -> bf16 WMMA
// (v_wmma_f32_16x16x32_bf16, fp32 accumulate), fp32 everywhere else.
// Weights are pre-packed to bf16 once (kernel 0) so GEMM inner loops do no
// format conversion; all LDS tiles use pitch 68 dwords (16B aligned, bank-
// conflict-free) so A-frags are 2x ds_load_b128.
// ---------------------------------------------------------------------------

typedef __bf16  v16bf    __attribute__((ext_vector_type(16)));
typedef __bf16  bf16x2_t __attribute__((ext_vector_type(2)));
typedef float   v8f      __attribute__((ext_vector_type(8)));

union FragBF { unsigned u[8]; uint4 q[2]; v16bf v; };

#define DEVI static __device__ __forceinline__

DEVI unsigned pkbf(float a, float b){
#if __has_builtin(__builtin_amdgcn_cvt_pk_bf16_f32)
  union { bf16x2_t v; unsigned u; } cv;
  cv.v = __builtin_amdgcn_cvt_pk_bf16_f32(a, b);
  return cv.u;
#else
  unsigned ua = __float_as_uint(a), ub = __float_as_uint(b);
  ua += 0x7fffu + ((ua >> 16) & 1u);
  ub += 0x7fffu + ((ub >> 16) & 1u);
  return (ua >> 16) | (ub & 0xffff0000u);
#endif
}
DEVI unsigned short f2bf(float f){ return (unsigned short)(pkbf(f, 0.f) & 0xffffu); }
DEVI float bf2f(unsigned short h){ return __uint_as_float(((unsigned)h) << 16); }

// A-frag: 16x32 bf16, rows = pixels, from LDS packed-pair tile (pitch dwords,
// pitch % 4 == 0 so both halves are 16B aligned -> 2x ds_load_b128).
// ISA layout: lanes 0-15 (row M=lane) hold K pairs {0..3}U{8..11},
// lanes 16-31 (row M=lane-16) hold pairs {4..7}U{12..15}.
DEVI FragBF load_a(const unsigned* t, int m0, int kk, int lane, int pitch){
  FragBF f;
  int l = lane & 15, hi = lane >> 4;
  const unsigned* row = t + (m0 + l) * pitch + kk * 16 + hi * 4;
  f.q[0] = *(const uint4*)row;
  f.q[1] = *(const uint4*)(row + 8);
  return f;
}

// B-frag: 32x16 bf16 = W^T slab from the pre-packed bf16 weight buffer
// (wbuf[row][64] dword pairs).  Same K striping as A: lanes 0-15 hold K pairs
// {0..3}U{8..11}, lanes 16-31 hold pairs {4..7}U{12..15}; N = lane%16.
DEVI FragBF load_b(const unsigned* __restrict__ wbuf, int row0, int kk, int lane){
  FragBF f;
  const unsigned* r = wbuf + (size_t)(row0 + (lane & 15)) * 64 + kk * 16 + (lane >> 4) * 4;
  f.q[0] = ((const uint4*)r)[0];
  f.q[1] = ((const uint4*)(r + 8))[0];
  return f;
}

DEVI v8f wmma_bf16(FragBF a, FragBF b, v8f c){
  return __builtin_amdgcn_wmma_f32_16x16x32_bf16(false, a.v, false, b.v,
                                                 (short)0, c, false, false);
}

// wbuf row bases: Wq=0, Wk=64, Wvp=128, Wvi=256, Wout=384 (512 rows x 64 dw)
#define RQ  0
#define RK  64
#define RVP 128
#define RVI 256
#define RO  384

// ---------------------------------------------------------------------------
// Kernel 0: one-shot fp32 -> bf16 weight pack (128 KB, stays in L2).
// ---------------------------------------------------------------------------
__global__ __launch_bounds__(256)
void pack_weights_kernel(const float* __restrict__ Wq,  const float* __restrict__ Wk,
                         const float* __restrict__ Wvp, const float* __restrict__ Wvi,
                         const float* __restrict__ Wout, unsigned* __restrict__ wbuf)
{
  int idx = blockIdx.x * 256 + threadIdx.x;   // 8192 threads, 8 channels each
  int row = idx >> 4, c8 = idx & 15;
  const float* src; int r;
  if      (row < 64)  { src = Wq;   r = row;       }
  else if (row < 128) { src = Wk;   r = row - 64;  }
  else if (row < 256) { src = Wvp;  r = row - 128; }
  else if (row < 384) { src = Wvi;  r = row - 256; }
  else                { src = Wout; r = row - 384; }
  const float4* s4 = (const float4*)(src + (size_t)r * 128 + c8 * 8);
  float4 a = s4[0], b = s4[1];
  uint4 o;
  o.x = pkbf(a.x, a.y); o.y = pkbf(a.z, a.w);
  o.z = pkbf(b.x, b.y); o.w = pkbf(b.z, b.w);
  *(uint4*)(wbuf + (size_t)row * 64 + c8 * 4) = o;
}

// ---------------------------------------------------------------------------
// Kernel 1: coarse grid.  kvbuf[b][p][0:64]=k, [64:192]=v (pixel-major so the
// attention kernel reads contiguous 768B per window tap).
// ---------------------------------------------------------------------------
__global__ __launch_bounds__(256)
void kv_coarse_kernel(const float* __restrict__ h_prev,
                      const unsigned* __restrict__ wbuf,
                      const float* __restrict__ bk, const float* __restrict__ bvp,
                      float* __restrict__ kvbuf)
{
  __shared__ unsigned s_h[128 * 68];          // 128 px x 64 ch-pairs, pitch 68

  const int b   = blockIdx.y;
  const int p0  = blockIdx.x * 128;           // 128 consecutive coarse pixels
  const int tid = threadIdx.x;

  for (int idx = tid; idx < 2048; idx += 256){   // float4 along pixels
    int c2 = idx >> 5, qd = idx & 31;
    size_t g = ((size_t)(b * 128 + 2 * c2)) * 32768 + p0 + qd * 4;
    float4 a = *(const float4*)(h_prev + g);
    float4 c = *(const float4*)(h_prev + g + 32768);
    unsigned* d = &s_h[(qd * 4) * 68 + c2];
    d[0]   = pkbf(a.x, c.x); d[68]  = pkbf(a.y, c.y);
    d[136] = pkbf(a.z, c.z); d[204] = pkbf(a.w, c.w);
  }
  __syncthreads();

  const int w = tid >> 5, lane = tid & 31;
  const int l = lane & 15, hi = lane >> 4;

  FragBF a[4];
#pragma unroll
  for (int kk = 0; kk < 4; ++kk) a[kk] = load_a(s_h, w * 16, kk, lane, 68);

  const size_t pb = (size_t)(b * 32768 + p0 + w * 16 + 8 * hi);

#pragma unroll
  for (int nt = 0; nt < 4; ++nt){             // k channels
    const int o0 = nt * 16;
    v8f acc = {};
#pragma unroll
    for (int kk = 0; kk < 4; ++kk) acc = wmma_bf16(a[kk], load_b(wbuf, RK + o0, kk, lane), acc);
    const float bias = bk[o0 + l];
#pragma unroll
    for (int r = 0; r < 8; ++r)
      kvbuf[(pb + r) * 192 + o0 + l] = acc[r] + bias;
  }
#pragma unroll
  for (int nt = 0; nt < 8; ++nt){             // v channels
    const int o0 = nt * 16;
    v8f acc = {};
#pragma unroll
    for (int kk = 0; kk < 4; ++kk) acc = wmma_bf16(a[kk], load_b(wbuf, RVP + o0, kk, lane), acc);
    const float bias = bvp[o0 + l];
#pragma unroll
    for (int r = 0; r < 8; ++r)
      kvbuf[(pb + r) * 192 + 64 + o0 + l] = acc[r] + bias;
  }
}

// ---------------------------------------------------------------------------
// Kernel 2: fused fine grid.  Tile = 8x16 fine pixels (128 px) per 256-thread
// block; coarse halo = 6x10 cells.  q GEMM -> softmax attention -> v_init GEMM
// (+x) -> out GEMM, all through LDS with one read of h_init.
// ---------------------------------------------------------------------------
__global__ __launch_bounds__(256)
void fused_attn_kernel(const float* __restrict__ h_init,
                       const unsigned* __restrict__ wbuf,
                       const float* __restrict__ bq,  const float* __restrict__ bvi,
                       const float* __restrict__ bout,
                       const float* __restrict__ kvbuf,
                       float* __restrict__ out)
{
  __shared__ unsigned s_h[128 * 68];          // h tile bf16 pairs; reused for t
  __shared__ float    s_q[128 * 68];          // q fp32; reused for x (bf16 pairs)
  __shared__ float    s_kv[60 * 192];         // 6x10 coarse halo, [cell][192]

  const int b   = blockIdx.z;
  const int fy0 = blockIdx.y * 8, fx0 = blockIdx.x * 16;
  const int cy0 = fy0 >> 1,       cx0 = fx0 >> 1;
  const int tid = threadIdx.x;

  // ---- stage h_init tile (read once, convert to bf16 pairs) ----
  for (int idx = tid; idx < 2048; idx += 256){
    int c2 = idx >> 5, qd = idx & 31;
    int p = qd * 4;
    int rr = p >> 4, cc = p & 15;
    size_t g = ((size_t)(b * 128 + 2 * c2)) * 131072 + (size_t)(fy0 + rr) * 512 + fx0 + cc;
    float4 a = *(const float4*)(h_init + g);
    float4 c = *(const float4*)(h_init + g + 131072);
    unsigned* d = &s_h[p * 68 + c2];
    d[0]   = pkbf(a.x, c.x); d[68]  = pkbf(a.y, c.y);
    d[136] = pkbf(a.z, c.z); d[204] = pkbf(a.w, c.w);
  }
  // ---- stage k/v halo, zero-filled OOB (== zero-padded unfold) ----
  for (int idx = tid; idx < 2880; idx += 256){
    int cell = idx / 48, j = idx - cell * 48;
    int hy = cell / 10, hx = cell - hy * 10;
    int cy = cy0 - 1 + hy, cx = cx0 - 1 + hx;
    float4 v = {0.f, 0.f, 0.f, 0.f};
    if (cy >= 0 && cy < 128 && cx >= 0 && cx < 256)
      v = *(const float4*)(kvbuf + ((size_t)(b * 32768 + cy * 256 + cx)) * 192 + j * 4);
    *(float4*)&s_kv[cell * 192 + j * 4] = v;
  }
  __syncthreads();

  const int w = tid >> 5, lane = tid & 31;
  const int l = lane & 15, hi = lane >> 4;

  // ---- q = Wq.h + bq : wave w owns M-tile w ----
  {
    FragBF a[4];
#pragma unroll
    for (int kk = 0; kk < 4; ++kk) a[kk] = load_a(s_h, w * 16, kk, lane, 68);
#pragma unroll
    for (int nt = 0; nt < 4; ++nt){
      const int o0 = nt * 16;
      v8f acc = {};
#pragma unroll
      for (int kk = 0; kk < 4; ++kk) acc = wmma_bf16(a[kk], load_b(wbuf, RQ + o0, kk, lane), acc);
      const float bias = bq[o0 + l];
#pragma unroll
      for (int r = 0; r < 8; ++r)
        s_q[(w * 16 + 8 * hi + r) * 68 + o0 + l] = acc[r] + bias;
    }
  }
  __syncthreads();

  // ---- attention: 2 threads per pixel (split 64-ch dot / 128-ch v sum) ----
  {
    const int p  = tid >> 1, hh = tid & 1;
    const int rr = p >> 4,   cc = p & 15;
    const int hy = (rr >> 1) + 1, hx = (cc >> 1) + 1;

    float qv[32];
#pragma unroll
    for (int j = 0; j < 8; ++j){
      float4 t4 = *(const float4*)&s_q[p * 68 + hh * 32 + j * 4];
      qv[4*j] = t4.x; qv[4*j+1] = t4.y; qv[4*j+2] = t4.z; qv[4*j+3] = t4.w;
    }

    float sc[9];
#pragma unroll
    for (int t = 0; t < 9; ++t){
      int dy = t / 3, dx = t - dy * 3;
      const float4* k4 = (const float4*)&s_kv[((hy + dy - 1) * 10 + (hx + dx - 1)) * 192 + hh * 32];
      float d = 0.f;
#pragma unroll
      for (int j = 0; j < 8; ++j){
        float4 kq = k4[j];
        d = fmaf(qv[4*j], kq.x, d); d = fmaf(qv[4*j+1], kq.y, d);
        d = fmaf(qv[4*j+2], kq.z, d); d = fmaf(qv[4*j+3], kq.w, d);
      }
      d += __shfl_xor(d, 1, 32);              // partner lane holds other 32 ch
      sc[t] = d;
    }
    float mx = sc[0];
#pragma unroll
    for (int t = 1; t < 9; ++t) mx = fmaxf(mx, sc[t]);
    float ssum = 0.f;
#pragma unroll
    for (int t = 0; t < 9; ++t){ sc[t] = __expf(sc[t] - mx); ssum += sc[t]; }
    const float inv = 1.f / ssum;

    float xacc[64];
#pragma unroll
    for (int c = 0; c < 64; ++c) xacc[c] = 0.f;
#pragma unroll
    for (int t = 0; t < 9; ++t){
      int dy = t / 3, dx = t - dy * 3;
      const float4* v4 = (const float4*)&s_kv[((hy + dy - 1) * 10 + (hx + dx - 1)) * 192 + 64 + hh * 64];
      const float wgt = sc[t] * inv;
#pragma unroll
      for (int j = 0; j < 16; ++j){
        float4 vv = v4[j];
        xacc[4*j]   = fmaf(wgt, vv.x, xacc[4*j]);
        xacc[4*j+1] = fmaf(wgt, vv.y, xacc[4*j+1]);
        xacc[4*j+2] = fmaf(wgt, vv.z, xacc[4*j+2]);
        xacc[4*j+3] = fmaf(wgt, vv.w, xacc[4*j+3]);
      }
    }
    // reuse s_q as x (bf16 pairs); each thread touches only its own row slice
    unsigned* s_x = (unsigned*)s_q;
#pragma unroll
    for (int j = 0; j < 8; ++j){
      uint4 o;
      o.x = pkbf(xacc[8*j],   xacc[8*j+1]); o.y = pkbf(xacc[8*j+2], xacc[8*j+3]);
      o.z = pkbf(xacc[8*j+4], xacc[8*j+5]); o.w = pkbf(xacc[8*j+6], xacc[8*j+7]);
      *(uint4*)&s_x[p * 68 + hh * 32 + j * 4] = o;
    }
  }
  __syncthreads();

  // ---- t = Wvi.h + bvi + x : keep t-frags in registers across rewrite ----
  const unsigned* s_x = (const unsigned*)s_q;
  v8f tf[8];
  {
    FragBF a[4];
#pragma unroll
    for (int kk = 0; kk < 4; ++kk) a[kk] = load_a(s_h, w * 16, kk, lane, 68);
#pragma unroll
    for (int nt = 0; nt < 8; ++nt){
      const int o0 = nt * 16;
      v8f acc = {};
#pragma unroll
      for (int kk = 0; kk < 4; ++kk) acc = wmma_bf16(a[kk], load_b(wbuf, RVI + o0, kk, lane), acc);
      const float bias = bvi[o0 + l];
      const int ch = o0 + l;
#pragma unroll
      for (int r = 0; r < 8; ++r){
        int m = w * 16 + 8 * hi + r;
        unsigned pr = s_x[m * 68 + (ch >> 1)];
        acc[r] += bias + bf2f((unsigned short)((ch & 1) ? (pr >> 16) : (pr & 0xffffu)));
      }
      tf[nt] = acc;
    }
  }
  __syncthreads();                            // all reads of s_h complete
  {
    unsigned short* s_t = (unsigned short*)s_h;   // t as bf16, pitch 136 halves
#pragma unroll
    for (int nt = 0; nt < 8; ++nt){
      const int ch = nt * 16 + l;
#pragma unroll
      for (int r = 0; r < 8; ++r)
        s_t[(w * 16 + 8 * hi + r) * 136 + ch] = f2bf(tf[nt][r]);
    }
  }
  __syncthreads();

  // ---- out = Wout.t + bout : per-lane 8 consecutive pixels -> 2x b128 store ----
  {
    FragBF a[4];
#pragma unroll
    for (int kk = 0; kk < 4; ++kk) a[kk] = load_a(s_h, w * 16, kk, lane, 68);
    const int m0r = w * 16 + 8 * hi;
    const int rr  = m0r >> 4, cc0 = m0r & 15;
#pragma unroll
    for (int nt = 0; nt < 8; ++nt){
      const int o0 = nt * 16;
      v8f acc = {};
#pragma unroll
      for (int kk = 0; kk < 4; ++kk) acc = wmma_bf16(a[kk], load_b(wbuf, RO + o0, kk, lane), acc);
      const float bias = bout[o0 + l];
      float* op = out + ((size_t)(b * 128 + o0 + l) * 256 + fy0 + rr) * 512 + fx0 + cc0;
      float4 s0 = {acc[0] + bias, acc[1] + bias, acc[2] + bias, acc[3] + bias};
      float4 s1 = {acc[4] + bias, acc[5] + bias, acc[6] + bias, acc[7] + bias};
      *(float4*)op       = s0;
      *(float4*)(op + 4) = s1;
    }
  }
}

// ---------------------------------------------------------------------------
extern "C" void kernel_launch(void* const* d_in, const int* in_sizes, int n_in,
                              void* d_out, int out_size, void* d_ws, size_t ws_size,
                              hipStream_t stream)
{
  const float* h_prev = (const float*)d_in[0];
  const float* h_init = (const float*)d_in[1];
  const float* Wq   = (const float*)d_in[2];
  const float* bq   = (const float*)d_in[3];
  const float* Wk   = (const float*)d_in[4];
  const float* bk   = (const float*)d_in[5];
  const float* Wvp  = (const float*)d_in[6];
  const float* bvp  = (const float*)d_in[7];
  const float* Wvi  = (const float*)d_in[8];
  const float* bvi  = (const float*)d_in[9];
  const float* Wout = (const float*)d_in[10];
  const float* bout = (const float*)d_in[11];
  float* out = (float*)d_out;

  unsigned* wbuf  = (unsigned*)d_ws;                       // 512*64*4 = 128 KB
  float*    kvbuf = (float*)((char*)d_ws + 512 * 64 * 4);  // 48 MB

  pack_weights_kernel<<<dim3(32), 256, 0, stream>>>(Wq, Wk, Wvp, Wvi, Wout, wbuf);
  kv_coarse_kernel<<<dim3(256, 2), 256, 0, stream>>>(h_prev, wbuf, bk, bvp, kvbuf);
  fused_attn_kernel<<<dim3(32, 32, 2), 256, 0, stream>>>(h_init, wbuf, bq, bvi, bout,
                                                         kvbuf, out);
}
